// DiffusionTSF_53034256171456
// MI455X (gfx1250) — compile-verified, matrix operand order checked by reading, not run on hardware
//
#include <hip/hip_runtime.h>
#include <hip/hip_bf16.h>
#include <stdint.h>

#define Bn 256
#define Hn 512
#define Tn 720
#define Kb 5
#define RAD 10
#define Wn 21              // 2*RAD+1
#define NCAND (Kb * Wn)    // 105
#define JUMPP 1.0f
#define EPSF 1e-8f
#define NEG_INF (-__builtin_inff())
#define WIN 128            // prefetch window rows (4 async ops of 32 lanes)
#define NOPS 4

typedef __attribute__((ext_vector_type(16))) _Float16 v16h;
typedef __attribute__((ext_vector_type(8)))  float    v8f;

// ---------------------------------------------------------------------------
// Phase 1: colsum via WMMA.  One wave per (b, 16-wide t-tile).
// A (16x32 f16) = clamped adjacent-row diffs, M = t-row, K = h chunk.
// B (32x16 f16) = ones  =>  D[m,n] = sum_k A[m,k], accumulated f32 over H.
// Two independent accumulators break the WMMA RAW chain; all loads are
// unconditional (no exec-mask branches, one wait per 36-load batch).
// Stores invs[b*T+t] = 1 / max(colsum, EPS).
// ---------------------------------------------------------------------------
__global__ __launch_bounds__(32) void colsum_wmma_kernel(const float* __restrict__ cdf,
                                                         float* __restrict__ invs) {
    const int wid  = blockIdx.x;            // 0 .. B*(T/16)-1
    const int b    = wid / (Tn / 16);
    const int tt   = wid - b * (Tn / 16);
    const int t0   = tt * 16;
    const int lane = threadIdx.x;
    const int m    = lane & 15;             // t-row this lane supplies (A-matrix M)
    const int kb   = (lane >> 4) << 3;      // 0 or 8: which K-half group
    // base pointer already includes the lane's kb row offset; relative row r
    // within an iteration maps to absolute row  h0 + kb + r.
    const float* ptr = cdf + (size_t)b * Hn * Tn + (size_t)kb * Tn + (t0 + m);

    v16h ones;
#pragma unroll
    for (int i = 0; i < 16; ++i) ones[i] = (_Float16)1.0f;

    v8f acc0 = {}, acc1 = {};

    // main: rows 0..447, max relative row = 56, all loads guard-free
#pragma unroll 1
    for (int h0 = 0; h0 < 448; h0 += 64) {
        float va0[9], vb0[9], va1[9], vb1[9];
#pragma unroll
        for (int j = 0; j < 9; ++j) {
            va0[j] = ptr[(size_t)(j)      * Tn];
            vb0[j] = ptr[(size_t)(16 + j) * Tn];
            va1[j] = ptr[(size_t)(32 + j) * Tn];
            vb1[j] = ptr[(size_t)(48 + j) * Tn];
        }
        v16h a0, a1;
#pragma unroll
        for (int j = 0; j < 8; ++j) {
            a0[j]     = (_Float16)fmaxf(va0[j] - va0[j + 1], 0.f);
            a0[8 + j] = (_Float16)fmaxf(vb0[j] - vb0[j + 1], 0.f);
            a1[j]     = (_Float16)fmaxf(va1[j] - va1[j + 1], 0.f);
            a1[8 + j] = (_Float16)fmaxf(vb1[j] - vb1[j + 1], 0.f);
        }
        acc0 = __builtin_amdgcn_wmma_f32_16x16x32_f16(false, a0, false, ones,
                                                      (short)0, acc0, false, false);
        acc1 = __builtin_amdgcn_wmma_f32_16x16x32_f16(false, a1, false, ones,
                                                      (short)0, acc1, false, false);
        ptr += (size_t)64 * Tn;
    }

    // tail: rows 448..511.  Only vb1 can reach absolute row 512 (kb==8, j==8):
    // clamp the index (cndmask, no branch); only a1[15] of the kb==8 half is the
    // forced-zero last-row diff.
    {
        float va0[9], vb0[9], va1[9], vb1[9];
#pragma unroll
        for (int j = 0; j < 9; ++j) {
            va0[j] = ptr[(size_t)(j)      * Tn];
            vb0[j] = ptr[(size_t)(16 + j) * Tn];
            va1[j] = ptr[(size_t)(32 + j) * Tn];
            int rc = min(48 + j, 63 - kb);        // abs row = 448 + kb + rc <= 511
            vb1[j] = ptr[(size_t)rc * Tn];
        }
        v16h a0, a1;
#pragma unroll
        for (int j = 0; j < 8; ++j) {
            a0[j]     = (_Float16)fmaxf(va0[j] - va0[j + 1], 0.f);
            a0[8 + j] = (_Float16)fmaxf(vb0[j] - vb0[j + 1], 0.f);
            a1[j]     = (_Float16)fmaxf(va1[j] - va1[j + 1], 0.f);
            float d   = fmaxf(vb1[j] - vb1[j + 1], 0.f);
            if (j == 7) d = (kb == 8) ? 0.f : d;  // absolute row 511 -> pdf = 0
            a1[8 + j] = (_Float16)d;
        }
        acc0 = __builtin_amdgcn_wmma_f32_16x16x32_f16(false, a0, false, ones,
                                                      (short)0, acc0, false, false);
        acc1 = __builtin_amdgcn_wmma_f32_16x16x32_f16(false, a1, false, ones,
                                                      (short)0, acc1, false, false);
    }

    v8f acc = acc0 + acc1;
    // D layout: lanes 0-15 VGPR r = D[r, lane]; lanes 16-31 VGPR r = D[r+8, lane-16].
    // All columns identical (B = ones) -> lane 0 holds colsum(t0+0..7), lane 16 holds (t0+8..15).
    if (lane == 0 || lane == 16) {
        const int mbase = (lane >> 4) << 3;
#pragma unroll
        for (int r = 0; r < 8; ++r)
            invs[b * Tn + t0 + mbase + r] = 1.f / fmaxf(acc[r], EPSF);
    }
}

// ---------------------------------------------------------------------------
// Phase 2: beam search.  One wave32 per batch element.
// ---------------------------------------------------------------------------
__device__ __forceinline__ void amax2(float& s, int& c, float s2, int c2) {
    // higher score wins; ties -> lower index (matches jax top_k / argmax)
    if (s2 > s || (s2 == s && c2 < c)) { s = s2; c = c2; }
}

__global__ __launch_bounds__(32) void beam_kernel(const float* __restrict__ cdf,
                                                  const float* __restrict__ binc,
                                                  const float* __restrict__ invs,
                                                  float* __restrict__ out) {
    __shared__ float    colbuf[2][WIN];     // double-buffered prefetch window
    __shared__ uint16_t bp[Tn * Kb];        // backpointers: (parent<<9)|pos
    __shared__ uint16_t postrace[Tn];
    __shared__ int      bs_pos[2][Kb];
    __shared__ float    bs_score[2][Kb];

    const int    lane  = threadIdx.x;
    const int    b     = blockIdx.x;
    const size_t bbase = (size_t)b * Hn * Tn;

    // ---- init step t = 0: top-5 of log_pdf[:,0] over H ----
    const float inv0 = invs[b * Tn];
    float s16[16];
#pragma unroll
    for (int i = 0; i < 16; ++i) {
        int h = lane + (i << 5);
        float d;
        if (h == Hn - 1) d = 0.f;
        else {
            float a = cdf[bbase + (size_t)h * Tn];
            float c = cdf[bbase + (size_t)(h + 1) * Tn];
            d = fmaxf(a - c, 0.f);
        }
        s16[i] = __logf(fmaxf(d * inv0, EPSF));
    }
    int winh[Kb]; float winsv[Kb];
#pragma unroll
    for (int sel = 0; sel < Kb; ++sel) {
        float ls = NEG_INF; int lc = 0x7FFFFFFF;
#pragma unroll
        for (int i = 0; i < 16; ++i) {
            int h = lane + (i << 5);
            bool skip = false;
#pragma unroll
            for (int q = 0; q < Kb; ++q)
                if (q < sel && h == winh[q]) skip = true;
            if (!skip) amax2(ls, lc, s16[i], h);
        }
#pragma unroll
        for (int off = 16; off >= 1; off >>= 1) {
            float os = __shfl_xor(ls, off, 32);
            int   oc = __shfl_xor(lc, off, 32);
            amax2(ls, lc, os, oc);
        }
        winh[sel] = lc; winsv[sel] = ls;
    }
    if (lane == 0) {
#pragma unroll
        for (int sel = 0; sel < Kb; ++sel) {
            bs_pos[0][sel]   = winh[sel];
            bs_score[0][sel] = winsv[sel];
            bp[sel]          = (uint16_t)winh[sel];
        }
    }
    __syncthreads();

    int p = 0, cur = 0;

    // ---- prime: async prefetch of column 1 window into colbuf[0] ----
    int mn = bs_pos[0][0];
#pragma unroll
    for (int k = 1; k < Kb; ++k) mn = min(mn, bs_pos[0][k]);
    int lo_cur = max(0, mn - 2 * RAD);
#pragma unroll
    for (int i = 0; i < NOPS; ++i) {
        int row = min(lo_cur + lane + (i << 5), Hn - 1);
        uint64_t ga = (uint64_t)(uintptr_t)(cdf + bbase + (size_t)row * Tn + 1);
        uint32_t la = (uint32_t)(uintptr_t)(&colbuf[cur][lane + (i << 5)]);
        asm volatile("global_load_async_to_lds_b32 %0, %1, off"
                     :: "v"(la), "v"(ga) : "memory");
    }

    // ---- main DP over t = 1 .. T-1 ----
    for (int t = 1; t < Tn; ++t) {
        const int nxt = cur ^ 1;

        // window for column t+1 derived from P(t-1): candidates(t+1) ⊂ P(t-1) ± 20
        int mn2 = bs_pos[p][0];
#pragma unroll
        for (int k = 1; k < Kb; ++k) mn2 = min(mn2, bs_pos[p][k]);
        const int lo_n  = max(0, mn2 - 2 * RAD);
        const int tcol  = min(t + 1, Tn - 1);
#pragma unroll
        for (int i = 0; i < NOPS; ++i) {
            int row = min(lo_n + lane + (i << 5), Hn - 1);
            uint64_t ga = (uint64_t)(uintptr_t)(cdf + bbase + (size_t)row * Tn + tcol);
            uint32_t la = (uint32_t)(uintptr_t)(&colbuf[nxt][lane + (i << 5)]);
            asm volatile("global_load_async_to_lds_b32 %0, %1, off"
                         :: "v"(la), "v"(ga) : "memory");
        }
        // allow the NOPS just issued to remain outstanding; column t's window is then complete
        asm volatile("s_wait_asynccnt 4" ::: "memory");

        const float invt = invs[b * Tn + t];

        // 105 candidates spread over 32 lanes (4 slots each)
        float cs[4];
#pragma unroll
        for (int j = 0; j < 4; ++j) {
            const int cand = lane + (j << 5);
            float sc = NEG_INF;
            if (cand < NCAND) {
                const int k   = cand / Wn;
                const int w   = cand - k * Wn;
                const int pos = bs_pos[p][k] + (w - RAD);
                if (pos >= 0 && pos < Hn) {
                    float d;
                    if (pos == Hn - 1) d = 0.f;
                    else {
                        const int slot = pos - lo_cur;
                        float a, c;
                        if (slot >= 0 && slot + 1 < WIN) {
                            a = colbuf[cur][slot];
                            c = colbuf[cur][slot + 1];
                        } else { // rare wide-spread fallback
                            a = cdf[bbase + (size_t)pos * Tn + t];
                            c = cdf[bbase + (size_t)(pos + 1) * Tn + t];
                        }
                        d = fmaxf(a - c, 0.f);
                    }
                    const float lp = __logf(fmaxf(d * invt, EPSF));
                    sc = bs_score[p][k] + lp - JUMPP * fabsf((float)(w - RAD));
                }
            }
            cs[j] = sc;
        }

        // top-5 of 105 via 5 wave32 butterfly argmax passes
        int wc[Kb]; float ws[Kb];
#pragma unroll
        for (int sel = 0; sel < Kb; ++sel) {
            float ls = NEG_INF; int lc = 0x7FFFFFFF;
#pragma unroll
            for (int j = 0; j < 4; ++j) {
                const int cand = lane + (j << 5);
                bool skip = false;
#pragma unroll
                for (int q = 0; q < Kb; ++q)
                    if (q < sel && cand == wc[q]) skip = true;
                if (!skip) amax2(ls, lc, cs[j], cand);
            }
#pragma unroll
            for (int off = 16; off >= 1; off >>= 1) {
                float os = __shfl_xor(ls, off, 32);
                int   oc = __shfl_xor(lc, off, 32);
                amax2(ls, lc, os, oc);
            }
            wc[sel] = lc; ws[sel] = ls;
        }

        // commit new beams + backpointers (ping-pong beam state)
        if (lane == 0) {
#pragma unroll
            for (int sel = 0; sel < Kb; ++sel) {
                const int c    = wc[sel];
                const int k    = c / Wn;
                const int w    = c - k * Wn;
                const int npos = bs_pos[p][k] + (w - RAD);
                bs_pos[p ^ 1][sel]   = npos;
                bs_score[p ^ 1][sel] = ws[sel];
                bp[t * Kb + sel]     = (uint16_t)((k << 9) | npos);
            }
        }
        __syncthreads();
        p ^= 1;
        lo_cur = lo_n;
        cur = nxt;
    }

    asm volatile("s_wait_asynccnt 0" ::: "memory");

    // ---- traceback + emit ----
    if (lane == 0) {
        int best = 0; float bsc = bs_score[p][0];
#pragma unroll
        for (int k = 1; k < Kb; ++k) {
            float v = bs_score[p][k];
            if (v > bsc) { bsc = v; best = k; }
        }
        int curk = best;
        for (int t = Tn - 1; t >= 1; --t) {
            uint16_t e = bp[t * Kb + curk];
            postrace[t] = (uint16_t)(e & 511);
            curk = e >> 9;
        }
        postrace[0] = (uint16_t)(bp[curk] & 511);
    }
    __syncthreads();
    for (int t = lane; t < Tn; t += 32)
        out[(size_t)b * Tn + t] = binc[postrace[t]];
}

// ---------------------------------------------------------------------------
extern "C" void kernel_launch(void* const* d_in, const int* in_sizes, int n_in,
                              void* d_out, int out_size, void* d_ws, size_t ws_size,
                              hipStream_t stream) {
    (void)in_sizes; (void)n_in; (void)out_size; (void)ws_size;
    const float* cdf  = (const float*)d_in[0];   // (B, H, T) f32
    const float* binc = (const float*)d_in[1];   // (H,) f32
    float*       out  = (float*)d_out;           // (B, T) f32
    float*       invs = (float*)d_ws;            // B*T f32 scratch (737 KB)

    colsum_wmma_kernel<<<Bn * (Tn / 16), 32, 0, stream>>>(cdf, invs);
    beam_kernel<<<Bn, 32, 0, stream>>>(cdf, binc, invs, out);
}
